// PlacementGraphEncoder_49228915146780
// MI455X (gfx1250) — matrix operand order repaired and checked by reading, not compiled, not run on hardware
//
#include <hip/hip_runtime.h>
#include <hip/hip_bf16.h>

// ---------------------------------------------------------------------------
// CDNA5 / gfx1250 GNN encoder. All matmuls use V_WMMA_F32_16X16X32_F16
// (f16 A/B, f32 accumulate), wave32. Edge kernel processes TWO 16-edge
// sub-tiles per wave so each weight (B) fragment is shared by two WMMAs.
// ---------------------------------------------------------------------------

typedef __attribute__((ext_vector_type(8)))  _Float16 v8h;
typedef __attribute__((ext_vector_type(16))) _Float16 v16h;
typedef __attribute__((ext_vector_type(8)))  float    v8f;

#define D_HID 128

__device__ __forceinline__ void wave_lds_fence() {
  // LDS ops from one wave execute in order; we only need the data-return wait
  // before consuming values that round-tripped through LDS within this wave.
  __asm__ volatile("s_wait_dscnt 0" ::: "memory");
}

__device__ __forceinline__ float gelu_exact(float x) {
  return 0.5f * x * (1.0f + erff(x * 0.70710678118654752f));
}

// A-matrix fragment (16x32 f16, MxK) from a row-major f16 row pointer.
// Per ISA 7.12.2: lanes 0-15 row M=lane hold K {k0..k0+7, k0+16..k0+23};
// lanes 16-31 (same rows) hold K {k0+8..k0+15, k0+24..k0+31}.
__device__ __forceinline__ v16h load_a_f16(const _Float16* __restrict__ row,
                                           int k0, int lane) {
  const int koff = k0 + ((lane & 16) ? 8 : 0);
  v8h lo = *(const v8h*)(row + koff);
  v8h hi = *(const v8h*)(row + koff + 16);
  v16h a;
#pragma unroll
  for (int i = 0; i < 8; ++i) { a[i] = lo[i]; a[i + 8] = hi[i]; }
  return a;
}

// Same A fragment but sourced from an f32 row (convert on the fly).
__device__ __forceinline__ v16h load_a_f32(const float* __restrict__ row,
                                           int k0, int lane) {
  const int koff = k0 + ((lane & 16) ? 8 : 0);
  v16h a;
#pragma unroll
  for (int i = 0; i < 8; ++i) {
    a[i]     = (_Float16)row[koff + i];
    a[i + 8] = (_Float16)row[koff + 16 + i];
  }
  return a;
}

// B-matrix fragment (32x16 f16, KxN) from an N-major (transposed) weight
// buffer with row stride KP halfs. Lane L<16: col N=L, K {k0..k0+7, k0+16..k0+23};
// lane L>=16: col N=L-16, K {k0+8..k0+15, k0+24..k0+31}. Two aligned 16B loads per lane.
__device__ __forceinline__ v16h load_b_t(const _Float16* __restrict__ wt,
                                         int kp, int tile_n, int k0, int lane) {
  const int n = tile_n * 16 + (lane & 15);
  const int k = k0 + ((lane & 16) ? 8 : 0);
  const _Float16* p = wt + (size_t)n * kp + k;
  v8h lo = *(const v8h*)(p);
  v8h hi = *(const v8h*)(p + 16);
  v16h b;
#pragma unroll
  for (int i = 0; i < 8; ++i) { b[i] = lo[i]; b[i + 8] = hi[i]; }
  return b;
}

__device__ __forceinline__ v8f wmma16(v16h a, v16h b, v8f c) {
  return __builtin_amdgcn_wmma_f32_16x16x32_f16(false, a, false, b,
                                                (short)0, c, false, false);
}

// ---------------------------------------------------------------------------
// Weight transpose+convert: W (K x 128 row-major, f32) -> Wt (128 x KP, f16)
// ---------------------------------------------------------------------------
__global__ void transpose_f16_kernel(const float* __restrict__ W,
                                     _Float16* __restrict__ Wt,
                                     int K, int KP) {
  int idx = blockIdx.x * 256 + threadIdx.x;
  if (idx >= K * D_HID) return;
  int k = idx >> 7;
  int n = idx & 127;
  Wt[(size_t)n * KP + k] = (_Float16)W[idx];
}

// ---------------------------------------------------------------------------
// hidden = gelu(node_features @ W_in + b_in);  also keep f16 copy.
// One block (128 threads) per node.
// ---------------------------------------------------------------------------
__global__ void input_proj_kernel(const float* __restrict__ nf,
                                  const float* __restrict__ W,
                                  const float* __restrict__ b,
                                  float* __restrict__ hidden,
                                  _Float16* __restrict__ hidden_h) {
  __shared__ float xs[32];
  const int node = blockIdx.x;
  const int j = threadIdx.x;
  if (j < 32) xs[j] = nf[(size_t)node * 32 + j];
  __syncthreads();
  float s = b[j];
#pragma unroll
  for (int k = 0; k < 32; ++k) s += xs[k] * W[k * D_HID + j];
  s = gelu_exact(s);
  hidden[(size_t)node * D_HID + j] = s;
  hidden_h[(size_t)node * D_HID + j] = (_Float16)s;
}

__global__ void fill_zero_kernel(float* __restrict__ p, size_t n) {
  size_t i = (size_t)blockIdx.x * blockDim.x + threadIdx.x;
  size_t stride = (size_t)gridDim.x * blockDim.x;
  for (; i < n; i += stride) p[i] = 0.0f;
}

// ---------------------------------------------------------------------------
// Edge MLP + scatter-add.  One wave32 per 32-edge tile pair (M=32):
//   stage1: [h[src] | h[dst] | ew] @ W1 (257x128) + b1, GELU   (128 WMMA)
//   stage2: m1 @ W2 (128x128) + b2                            (64 WMMA)
//   scatter: atomicAdd into agg[dst]
// Each weight B-fragment is loaded once and feeds both sub-tiles.
// ---------------------------------------------------------------------------
__global__ __launch_bounds__(256) void edge_mlp_kernel(
    const _Float16* __restrict__ hidden_h,
    const int* __restrict__ src, const int* __restrict__ dst,
    const float* __restrict__ ew,
    const _Float16* __restrict__ w1t,   // 128 x 272 (K padded from 257)
    const float* __restrict__ b1,
    const _Float16* __restrict__ w2t,   // 128 x 128
    const float* __restrict__ b2,
    float* __restrict__ agg, int E_) {
  __shared__ __attribute__((aligned(32))) _Float16 lds[8][32 * D_HID];
  const int wave = threadIdx.x >> 5;
  const int lane = threadIdx.x & 31;
  const int pair = blockIdx.x * 8 + wave;
  const int e0 = pair * 32;
  if (e0 >= E_) return;

  // Warm the caches with this layer's weights once per block (cheap, async).
  {
    const char* wbase = (const char*)w1t;
    // w1t (69632B) + w2t (32768B) are contiguous per-layer? not guaranteed --
    // prefetch each separately.
    for (int off = threadIdx.x * 256; off < 128 * 272 * 2; off += 256 * 256)
      __builtin_prefetch(wbase + off, 0, 1);
    const char* wbase2 = (const char*)w2t;
    for (int off = threadIdx.x * 256; off < 128 * 128 * 2; off += 256 * 256)
      __builtin_prefetch(wbase2 + off, 0, 1);
  }

  const int m = lane & 15;
  const int mbase = (lane & 16) ? 8 : 0;
  const _Float16* srow0 = hidden_h + (size_t)src[e0 + m] * D_HID;
  const _Float16* drow0 = hidden_h + (size_t)dst[e0 + m] * D_HID;
  const _Float16* srow1 = hidden_h + (size_t)src[e0 + 16 + m] * D_HID;
  const _Float16* drow1 = hidden_h + (size_t)dst[e0 + 16 + m] * D_HID;

  v8f accA[8], accB[8];
#pragma unroll
  for (int t = 0; t < 8; ++t) {
    accA[t] = (v8f){0, 0, 0, 0, 0, 0, 0, 0};
    accB[t] = (v8f){0, 0, 0, 0, 0, 0, 0, 0};
  }

  // stage 1: src part (W1 rows 0..127)
#pragma unroll
  for (int kk = 0; kk < 4; ++kk) {
    v16h a0 = load_a_f16(srow0, kk * 32, lane);
    v16h a1 = load_a_f16(srow1, kk * 32, lane);
#pragma unroll
    for (int tn = 0; tn < 8; ++tn) {
      v16h b = load_b_t(w1t, 272, tn, kk * 32, lane);
      accA[tn] = wmma16(a0, b, accA[tn]);
      accB[tn] = wmma16(a1, b, accB[tn]);
    }
  }
  // stage 1: dst part (W1 rows 128..255)
#pragma unroll
  for (int kk = 0; kk < 4; ++kk) {
    v16h a0 = load_a_f16(drow0, kk * 32, lane);
    v16h a1 = load_a_f16(drow1, kk * 32, lane);
#pragma unroll
    for (int tn = 0; tn < 8; ++tn) {
      v16h b = load_b_t(w1t, 272, tn, 128 + kk * 32, lane);
      accA[tn] = wmma16(a0, b, accA[tn]);
      accB[tn] = wmma16(a1, b, accB[tn]);
    }
  }

  // rank-1 edge-weight term + bias + GELU, stage to LDS as f16.
  float ewv0[8], ewv1[8];
#pragma unroll
  for (int r = 0; r < 8; ++r) {
    ewv0[r] = ew[e0 + mbase + r];
    ewv1[r] = ew[e0 + 16 + mbase + r];
  }
#pragma unroll
  for (int tn = 0; tn < 8; ++tn) {
    const int n = tn * 16 + (lane & 15);
    const float wl = (float)w1t[(size_t)n * 272 + 256];  // W1 row 256
    const float bb = b1[n];
#pragma unroll
    for (int r = 0; r < 8; ++r) {
      float x0 = accA[tn][r] + ewv0[r] * wl + bb;
      float x1 = accB[tn][r] + ewv1[r] * wl + bb;
      lds[wave][(mbase + r) * D_HID + n] = (_Float16)gelu_exact(x0);
      lds[wave][(16 + mbase + r) * D_HID + n] = (_Float16)gelu_exact(x1);
    }
  }
  wave_lds_fence();

  // stage 2
  v8f acc2A[8], acc2B[8];
#pragma unroll
  for (int t = 0; t < 8; ++t) {
    acc2A[t] = (v8f){0, 0, 0, 0, 0, 0, 0, 0};
    acc2B[t] = (v8f){0, 0, 0, 0, 0, 0, 0, 0};
  }
  const _Float16* mrow0 = &lds[wave][m * D_HID];
  const _Float16* mrow1 = &lds[wave][(16 + m) * D_HID];
#pragma unroll
  for (int kk = 0; kk < 4; ++kk) {
    v16h a0 = load_a_f16(mrow0, kk * 32, lane);
    v16h a1 = load_a_f16(mrow1, kk * 32, lane);
#pragma unroll
    for (int tn = 0; tn < 8; ++tn) {
      v16h b = load_b_t(w2t, 128, tn, kk * 32, lane);
      acc2A[tn] = wmma16(a0, b, acc2A[tn]);
      acc2B[tn] = wmma16(a1, b, acc2B[tn]);
    }
  }

  // bias + scatter-add into agg[dst] (dst rows hoisted to registers)
  int dn0[8], dn1[8];
#pragma unroll
  for (int r = 0; r < 8; ++r) {
    dn0[r] = dst[e0 + mbase + r];
    dn1[r] = dst[e0 + 16 + mbase + r];
  }
#pragma unroll
  for (int tn = 0; tn < 8; ++tn) {
    const int n = tn * 16 + (lane & 15);
    const float bb = b2[n];
#pragma unroll
    for (int r = 0; r < 8; ++r) {
      atomicAdd(&agg[(size_t)dn0[r] * D_HID + n], acc2A[tn][r] + bb);
      atomicAdd(&agg[(size_t)dn1[r] * D_HID + n], acc2B[tn][r] + bb);
    }
  }
}

// ---------------------------------------------------------------------------
// Node update: cat=[hidden|agg] -> MLP -> residual -> LayerNorm.
// One wave32 per 16-node tile.
// ---------------------------------------------------------------------------
__global__ __launch_bounds__(256) void node_update_kernel(
    float* __restrict__ hidden, _Float16* __restrict__ hidden_h,
    const float* __restrict__ agg,
    const _Float16* __restrict__ w1t,   // 128 x 256
    const float* __restrict__ b1,
    const _Float16* __restrict__ w2t,   // 128 x 128
    const float* __restrict__ b2,
    const float* __restrict__ gamma, const float* __restrict__ beta,
    int Nn) {
  __shared__ __attribute__((aligned(32))) float ldsb[8][16 * D_HID];
  const int wave = threadIdx.x >> 5;
  const int lane = threadIdx.x & 31;
  const int tile = blockIdx.x * 8 + wave;
  const int n0 = tile * 16;
  if (n0 >= Nn) return;

  const int m = lane & 15;
  const int mbase = (lane & 16) ? 8 : 0;
  const _Float16* hrow = hidden_h + (size_t)(n0 + m) * D_HID;
  const float* arow = agg + (size_t)(n0 + m) * D_HID;

  v8f acc[8];
#pragma unroll
  for (int t = 0; t < 8; ++t) acc[t] = (v8f){0, 0, 0, 0, 0, 0, 0, 0};

#pragma unroll
  for (int kk = 0; kk < 4; ++kk) {            // hidden part (K 0..127)
    v16h a = load_a_f16(hrow, kk * 32, lane);
#pragma unroll
    for (int tn = 0; tn < 8; ++tn)
      acc[tn] = wmma16(a, load_b_t(w1t, 256, tn, kk * 32, lane), acc[tn]);
  }
#pragma unroll
  for (int kk = 0; kk < 4; ++kk) {            // agg part (K 128..255)
    v16h a = load_a_f32(arow, kk * 32, lane);
#pragma unroll
    for (int tn = 0; tn < 8; ++tn)
      acc[tn] = wmma16(a, load_b_t(w1t, 256, tn, 128 + kk * 32, lane), acc[tn]);
  }

  // bias + GELU -> f16 staging (low half of ldsb row reused as f16)
  _Float16* ldsh = (_Float16*)&ldsb[wave][0];
#pragma unroll
  for (int tn = 0; tn < 8; ++tn) {
    const int n = tn * 16 + (lane & 15);
    const float bb = b1[n];
#pragma unroll
    for (int r = 0; r < 8; ++r)
      ldsh[(mbase + r) * D_HID + n] = (_Float16)gelu_exact(acc[tn][r] + bb);
  }
  wave_lds_fence();

  v8f acc2[8];
#pragma unroll
  for (int t = 0; t < 8; ++t) acc2[t] = (v8f){0, 0, 0, 0, 0, 0, 0, 0};
  const _Float16* mrow = ldsh + m * D_HID;
#pragma unroll
  for (int kk = 0; kk < 4; ++kk) {
    v16h a = load_a_f16(mrow, kk * 32, lane);
#pragma unroll
    for (int tn = 0; tn < 8; ++tn)
      acc2[tn] = wmma16(a, load_b_t(w2t, 128, tn, kk * 32, lane), acc2[tn]);
  }

  // updated (+ bias) to LDS as f32 for LayerNorm redistribution
#pragma unroll
  for (int tn = 0; tn < 8; ++tn) {
    const int n = tn * 16 + (lane & 15);
    const float bb = b2[n];
#pragma unroll
    for (int r = 0; r < 8; ++r)
      ldsb[wave][(mbase + r) * D_HID + n] = acc2[tn][r] + bb;
  }
  wave_lds_fence();

  // residual + LayerNorm: each lane owns 4 columns per row, wave32 reduce.
  for (int r = 0; r < 16; ++r) {
    const size_t rowo = (size_t)(n0 + r) * D_HID;
    float x[4];
    float s = 0.0f;
#pragma unroll
    for (int i = 0; i < 4; ++i) {
      const int c = lane * 4 + i;
      x[i] = hidden[rowo + c] + ldsb[wave][r * D_HID + c];
      s += x[i];
    }
#pragma unroll
    for (int off = 16; off > 0; off >>= 1) s += __shfl_xor(s, off);
    const float mu = s * (1.0f / 128.0f);
    float v = 0.0f;
#pragma unroll
    for (int i = 0; i < 4; ++i) { float d = x[i] - mu; v += d * d; }
#pragma unroll
    for (int off = 16; off > 0; off >>= 1) v += __shfl_xor(v, off);
    const float inv = rsqrtf(v * (1.0f / 128.0f) + 1e-5f);
#pragma unroll
    for (int i = 0; i < 4; ++i) {
      const int c = lane * 4 + i;
      const float y = (x[i] - mu) * inv * gamma[c] + beta[c];
      hidden[rowo + c] = y;
      hidden_h[rowo + c] = (_Float16)y;
    }
  }
}

// ---------------------------------------------------------------------------
// Mean pooling over graphs.
// ---------------------------------------------------------------------------
__global__ void pool_accum_kernel(const float* __restrict__ hidden,
                                  const int* __restrict__ batch,
                                  float* __restrict__ pool,
                                  float* __restrict__ cnt, int Nn) {
  int idx = blockIdx.x * 256 + threadIdx.x;
  if (idx >= Nn * D_HID) return;
  const int node = idx >> 7;
  const int c = idx & 127;
  const int g = batch[node];
  atomicAdd(&pool[(size_t)g * D_HID + c], hidden[idx]);
  if (c == 0) atomicAdd(&cnt[g], 1.0f);
}

__global__ void pool_finalize_kernel(const float* __restrict__ pool,
                                     const float* __restrict__ cnt,
                                     float* __restrict__ out, int G_) {
  int idx = blockIdx.x * 256 + threadIdx.x;
  if (idx >= G_ * D_HID) return;
  const int g = idx >> 7;
  float c = cnt[g];
  c = (c < 1.0f) ? 1.0f : c;
  out[idx] = pool[idx] / c;
}

// ---------------------------------------------------------------------------
// Host-side orchestration.
// ---------------------------------------------------------------------------
extern "C" void kernel_launch(void* const* d_in, const int* in_sizes, int n_in,
                              void* d_out, int out_size, void* d_ws,
                              size_t ws_size, hipStream_t stream) {
  const float* node_features = (const float*)d_in[0];
  const int*   edge_index    = (const int*)d_in[1];
  const float* edge_weight   = (const float*)d_in[2];
  const int*   batch_index   = (const int*)d_in[3];
  const float* W_in   = (const float*)d_in[5];
  const float* b_in   = (const float*)d_in[6];
  const float* edge_W1 = (const float*)d_in[7];
  const float* edge_b1 = (const float*)d_in[8];
  const float* edge_W2 = (const float*)d_in[9];
  const float* edge_b2 = (const float*)d_in[10];
  const float* upd_W1  = (const float*)d_in[11];
  const float* upd_b1  = (const float*)d_in[12];
  const float* upd_W2  = (const float*)d_in[13];
  const float* upd_b2  = (const float*)d_in[14];
  const float* gamma   = (const float*)d_in[15];
  const float* beta    = (const float*)d_in[16];

  const int N = in_sizes[0] / 32;
  const int E = in_sizes[1] / 2;
  const int G = in_sizes[4];
  const int L = 3;
  const int* src = edge_index;
  const int* dst = edge_index + E;

  // workspace carve-up (256B aligned slabs)
  char* wsp = (char*)d_ws;
  auto take = [&](size_t bytes) -> char* {
    char* p = wsp;
    wsp += (bytes + 255) & ~(size_t)255;
    return p;
  };
  _Float16* hidden_h = (_Float16*)take((size_t)N * D_HID * 2);
  float*    agg      = (float*)take((size_t)N * D_HID * 4);
  _Float16* eW1t     = (_Float16*)take((size_t)L * 128 * 272 * 2);
  _Float16* eW2t     = (_Float16*)take((size_t)L * 128 * 128 * 2);
  _Float16* uW1t     = (_Float16*)take((size_t)L * 128 * 256 * 2);
  _Float16* uW2t     = (_Float16*)take((size_t)L * 128 * 128 * 2);
  float*    pool     = (float*)take((size_t)G * D_HID * 4 + (size_t)G * 4);
  float*    cnt      = pool + (size_t)G * D_HID;

  float* hidden   = (float*)d_out;              // N x 128 region of output
  float* out_pool = hidden + (size_t)N * D_HID; // G x 128 region of output

  // 1) weight transpose/convert to f16
  for (int l = 0; l < L; ++l) {
    transpose_f16_kernel<<<(257 * 128 + 255) / 256, 256, 0, stream>>>(
        edge_W1 + (size_t)l * 257 * 128, eW1t + (size_t)l * 128 * 272, 257, 272);
    transpose_f16_kernel<<<(128 * 128 + 255) / 256, 256, 0, stream>>>(
        edge_W2 + (size_t)l * 128 * 128, eW2t + (size_t)l * 128 * 128, 128, 128);
    transpose_f16_kernel<<<(256 * 128 + 255) / 256, 256, 0, stream>>>(
        upd_W1 + (size_t)l * 256 * 128, uW1t + (size_t)l * 128 * 256, 256, 256);
    transpose_f16_kernel<<<(128 * 128 + 255) / 256, 256, 0, stream>>>(
        upd_W2 + (size_t)l * 128 * 128, uW2t + (size_t)l * 128 * 128, 128, 128);
  }

  // 2) input projection
  input_proj_kernel<<<N, 128, 0, stream>>>(node_features, W_in, b_in, hidden,
                                           hidden_h);

  // 3) message-passing layers
  const int epairs = E / 32;  // E divisible by 32
  const int ntiles = N / 16;  // N divisible by 16
  for (int l = 0; l < L; ++l) {
    fill_zero_kernel<<<1024, 256, 0, stream>>>(agg, (size_t)N * D_HID);
    edge_mlp_kernel<<<(epairs + 7) / 8, 256, 0, stream>>>(
        hidden_h, src, dst, edge_weight,
        eW1t + (size_t)l * 128 * 272, edge_b1 + l * 128,
        eW2t + (size_t)l * 128 * 128, edge_b2 + l * 128, agg, E);
    node_update_kernel<<<(ntiles + 7) / 8, 256, 0, stream>>>(
        hidden, hidden_h, agg,
        uW1t + (size_t)l * 128 * 256, upd_b1 + l * 128,
        uW2t + (size_t)l * 128 * 128, upd_b2 + l * 128,
        gamma + l * 128, beta + l * 128, N);
  }

  // 4) graph mean pooling
  fill_zero_kernel<<<((G * D_HID + G) + 255) / 256, 256, 0, stream>>>(
      pool, (size_t)(G * D_HID + G));
  pool_accum_kernel<<<(N * D_HID + 255) / 256, 256, 0, stream>>>(
      hidden, batch_index, pool, cnt, N);
  pool_finalize_kernel<<<(G * D_HID + 255) / 256, 256, 0, stream>>>(
      pool, cnt, out_pool, G);
}
